// MdeformConvBlock_81381040325399
// MI455X (gfx1250) — matrix operand order repaired and checked by reading, not compiled, not run on hardware
//
#include <hip/hip_runtime.h>

typedef float        v8f   __attribute__((ext_vector_type(8)));
typedef __bf16       v8bf  __attribute__((ext_vector_type(8)));
typedef __bf16       v16bf __attribute__((ext_vector_type(16)));
typedef unsigned int u32;
typedef u32          u32x4 __attribute__((ext_vector_type(4)));
typedef u32          u32x8 __attribute__((ext_vector_type(8)));

#define Bn    2
#define CIN   256
#define COUT  256
#define Hn    96
#define Wn    96
#define HWn   (Hn*Wn)
#define KKn   9
#define DGn   4
#define KDIM  (CIN*KKn)        // 2304
#define MTOT  (Bn*HWn)         // 18432
#define NKC   (KDIM/32)        // 72 K-chunks of 32 (K permuted: k' = t*256 + c)
#define NOFF  108
#define NOFFP 128
#define NPTS  (Bn*DGn*KKn*HWn) // 663552 sample points

__device__ __forceinline__ __bf16 f2bf(float f) {
  unsigned u = __builtin_bit_cast(unsigned, f);
  unsigned r = u + 0x7FFFu + ((u >> 16) & 1u);   // round-to-nearest-even
  return __builtin_bit_cast(__bf16, (unsigned short)(r >> 16));
}

// ---- Tensor Data Mover: 1-row 2D descriptor, global -> LDS copy of N dwords.
// D# per CDNA5 ISA ch.8: group0 = {count=1, lds_addr, global_addr, type=2},
// group1 = {data_size=4B, tensor_dim0=N, tensor_dim1=1, tile_dim0=N, tile_dim1=1,
//           dim0_stride=N}.  Issued by one wave; tracked with TENSORcnt.
__device__ __forceinline__ void tdm_copy_b(const void* gsrc, void* ldsdst,
                                           unsigned ndw) {
  unsigned long long ga = (unsigned long long)(size_t)gsrc;
  unsigned lds_off = (unsigned)(size_t)ldsdst;   // low 32 bits = LDS byte offset
  u32x4 g0;
  g0[0] = 1u;                                     // count=1, user mode
  g0[1] = lds_off;                                // lds_addr
  g0[2] = (u32)ga;                                // global_addr[31:0]
  g0[3] = (u32)((ga >> 32) & 0x1FFFFFFu) | (2u << 30);  // addr[56:32] | type=2
  u32x8 g1;
  g1[0] = 2u << 16;                               // data_size = 4 bytes
  g1[1] = (ndw & 0xFFFFu) << 16;                  // tensor_dim0[15:0]
  g1[2] = ((ndw >> 16) & 0xFFFFu) | (1u << 16);   // tensor_dim0[31:16] | tensor_dim1=1
  g1[3] = (ndw & 0xFFFFu) << 16;                  // tile_dim0
  g1[4] = 1u;                                     // tile_dim1 = 1, tile_dim2 = 0
  g1[5] = ndw;                                    // tensor_dim0_stride[31:0]
  g1[6] = 0u;
  g1[7] = 0u;
  asm volatile("tensor_load_to_lds %0, %1" :: "s"(g0), "s"(g1) : "memory");
}

// ---- Pack w_conv (Cout,Cin,3,3) fp32 -> bf16 WMMA B-fragment layout,
// permuted K: k' = t*256 + c. frag e = ((kc*16+nsub)*32+lane)*16 + (2v+p),
// holds B[k'][n], k' = kc*32 + 2v + 16*(lane>>4) + p, n = nsub*16 + (lane&15).
__global__ __launch_bounds__(256) void pack_wconv_k(const float* __restrict__ w,
                                                    __bf16* __restrict__ dst) {
  int e = blockIdx.x * 256 + threadIdx.x;
  if (e >= KDIM * COUT) return;
  int j    = e & 15;
  int lane = (e >> 4) & 31;
  int nsub = (e >> 9) & 15;
  int kc   = e >> 13;
  int kp = kc * 32 + (j & ~1) + 16 * (lane >> 4) + (j & 1);
  int t = kp >> 8, c = kp & 255;
  int n = nsub * 16 + (lane & 15);
  dst[e] = f2bf(w[(size_t)n * KDIM + c * 9 + t]);
}

__global__ __launch_bounds__(256) void pack_woff_k(const float* __restrict__ w,
                                                   __bf16* __restrict__ dst) {
  int e = blockIdx.x * 256 + threadIdx.x;
  if (e >= KDIM * NOFFP) return;
  int j    = e & 15;
  int lane = (e >> 4) & 31;
  int nsub = (e >> 9) & 7;
  int kc   = e >> 12;
  int kp = kc * 32 + (j & ~1) + 16 * (lane >> 4) + (j & 1);
  int t = kp >> 8, c = kp & 255;
  int n = nsub * 16 + (lane & 15);
  dst[e] = (n < NOFF) ? f2bf(w[(size_t)n * KDIM + c * 9 + t]) : f2bf(0.0f);
}

// ---- Offset conv as implicit GEMM (double-buffered, TDM-staged B) ----
__global__ __launch_bounds__(256) void offset_gemm_k(const float* __restrict__ x,
                                                     const __bf16* __restrict__ bw,
                                                     const float* __restrict__ bias,
                                                     float* __restrict__ om) {
  __shared__ __align__(32) __bf16 ldsA[2][128 * 32];
  __shared__ __align__(32) __bf16 ldsB[2][NOFFP * 32];
  const int tid  = threadIdx.x;
  const int lane = tid & 31, wave = tid >> 5;
  const int half = lane >> 4, l16 = lane & 15;
  const int m0 = blockIdx.x * 128;

  const int r = tid >> 1, kb = (tid & 1) * 16;   // A-build: 2 threads/row
  const int m  = m0 + r;
  const int bb = m / HWn, hw = m % HWn;
  const int h  = hw / Wn, w = hw % Wn;

  auto buildA = [&](int kc, __bf16* dst) {       // fixed tap per chunk
    int t = kc >> 3;
    int cbase = (kc & 7) * 32 + kb;
    int ty = t / 3, tx = t - ty * 3;
    int yy = h + ty - 1, xx = w + tx - 1;
    bool valid = (yy >= 0) && (yy < Hn) && (xx >= 0) && (xx < Wn);
    const float* base = x + (((size_t)bb * CIN + cbase) * HWn + yy * Wn + xx);
    #pragma unroll
    for (int i = 0; i < 16; ++i) {
      float v = valid ? base[i * HWn] : 0.f;
      dst[r * 32 + kb + i] = f2bf(v);
    }
  };

  v8f acc[8] = {};

  if (tid < 32) tdm_copy_b(bw, &ldsB[0][0], 2048);
  buildA(0, &ldsA[0][0]);

  for (int kc = 0; kc < NKC; ++kc) {
    __builtin_amdgcn_s_wait_tensorcnt(0);
    __syncthreads();
    const int cur = kc & 1;
    if (kc + 1 < NKC) {
      if (tid < 32)
        tdm_copy_b(bw + (size_t)(kc + 1) * 4096, &ldsB[cur ^ 1][0], 2048);
      buildA(kc + 1, &ldsA[cur ^ 1][0]);
    }
    const __bf16* curA = &ldsA[cur][0];
    const __bf16* curB = &ldsB[cur][0];
    v8bf a0 = *(const v8bf*)&curA[(wave * 16 + l16) * 32 + half * 16];
    v8bf a1 = *(const v8bf*)&curA[(wave * 16 + l16) * 32 + half * 16 + 8];
    v16bf a = __builtin_shufflevector(a0, a1, 0,1,2,3,4,5,6,7,8,9,10,11,12,13,14,15);
    v16bf bcur = *(const v16bf*)&curB[lane * 16];
    #pragma unroll
    for (int n = 0; n < 8; ++n) {
      v16bf bnext = bcur;
      if (n < 7) bnext = *(const v16bf*)&curB[((n + 1) * 32 + lane) * 16];
      acc[n] = __builtin_amdgcn_wmma_f32_16x16x32_bf16(false, a, false, bcur,
                                                       (short)0, acc[n], false, false);
      bcur = bnext;
    }
  }
  #pragma unroll
  for (int n = 0; n < 8; ++n) {
    int ncol = n * 16 + l16;
    float bv = (ncol < NOFF) ? bias[ncol] : 0.f;
    #pragma unroll
    for (int g = 0; g < 8; ++g) {
      int ml = wave * 16 + g + 8 * half;
      om[(size_t)(m0 + ml) * NOFFP + ncol] = acc[n][g] + bv;
    }
  }
}

// ---- Per sample point: sigmoid mask folded into 4 bilinear weights + offsets ----
__global__ __launch_bounds__(256) void precomp_k(const float* __restrict__ om,
    float* __restrict__ w00, float* __restrict__ w01,
    float* __restrict__ w10, float* __restrict__ w11,
    int* __restrict__ o00, int* __restrict__ o01,
    int* __restrict__ o10, int* __restrict__ o11) {
  int p = blockIdx.x * 256 + threadIdx.x;
  if (p >= NPTS) return;
  int hw = p % HWn; int q = p / HWn;
  int t = q % KKn;  q /= KKn;
  int dg = q % DGn; int bb = q / DGn;
  int h = hw / Wn, w = hw % Wn;
  const float* row = om + (size_t)(bb * HWn + hw) * NOFFP;
  int u = dg * KKn + t;
  float dy = row[2 * u], dx = row[2 * u + 1];
  float mask = 1.f / (1.f + __expf(-row[72 + u]));
  float yf = (float)(h + t / 3 - 1) + dy;
  float xf = (float)(w + t % 3 - 1) + dx;
  float y0f = floorf(yf), x0f = floorf(xf);
  float ly = yf - y0f, lx = xf - x0f;
  int y0 = (int)y0f, x0 = (int)x0f;
  int y1 = y0 + 1,   x1 = x0 + 1;
  bool vy0 = (y0 >= 0) && (y0 < Hn), vy1 = (y1 >= 0) && (y1 < Hn);
  bool vx0 = (x0 >= 0) && (x0 < Wn), vx1 = (x1 >= 0) && (x1 < Wn);
  int y0c = min(max(y0, 0), Hn - 1), y1c = min(max(y1, 0), Hn - 1);
  int x0c = min(max(x0, 0), Wn - 1), x1c = min(max(x1, 0), Wn - 1);
  w00[p] = (vy0 && vx0) ? (1.f - ly) * (1.f - lx) * mask : 0.f;
  w01[p] = (vy0 && vx1) ? (1.f - ly) * lx * mask : 0.f;
  w10[p] = (vy1 && vx0) ? ly * (1.f - lx) * mask : 0.f;
  w11[p] = (vy1 && vx1) ? ly * lx * mask : 0.f;
  o00[p] = y0c * Wn + x0c;  o01[p] = y0c * Wn + x1c;
  o10[p] = y1c * Wn + x0c;  o11[p] = y1c * Wn + x1c;
}

// ---- Main deformable conv: implicit GEMM, fused bilinear gather, ----
// ---- double-buffered LDS, TDM-staged weights.                     ----
__global__ __launch_bounds__(256) void main_gemm_k(const float* __restrict__ x,
    const __bf16* __restrict__ bw,
    const float* __restrict__ w00, const float* __restrict__ w01,
    const float* __restrict__ w10, const float* __restrict__ w11,
    const int* __restrict__ o00, const int* __restrict__ o01,
    const int* __restrict__ o10, const int* __restrict__ o11,
    float* __restrict__ out) {
  __shared__ __align__(32) __bf16 ldsA[2][128 * 32];
  __shared__ __align__(32) __bf16 ldsB[2][256 * 32];
  const int tid  = threadIdx.x;
  const int lane = tid & 31, wave = tid >> 5;
  const int half = lane >> 4, l16 = lane & 15;
  const int m0 = blockIdx.x * 128;
  const int r = tid >> 1, kb = (tid & 1) * 16;
  const int m  = m0 + r;
  const int bb = m / HWn, hw = m % HWn;

  // Fixed tap per chunk -> one sample point (weights+offsets) per 16 channels.
  auto buildA = [&](int kc, __bf16* dst) {
    int t = kc >> 3;
    int cbase = (kc & 7) * 32 + kb;       // 16 consecutive channels, one group
    int dg = cbase >> 6;
    int p = ((bb * DGn + dg) * KKn + t) * HWn + hw;
    float f00 = w00[p], f01 = w01[p], f10 = w10[p], f11 = w11[p];
    int q00 = o00[p], q01 = o01[p], q10 = o10[p], q11 = o11[p];
    const float* xa = x + ((size_t)bb * CIN + cbase) * HWn;
    #pragma unroll
    for (int i = 0; i < 16; ++i) {
      int off = i * HWn;
      float s = f00 * xa[q00 + off] + f01 * xa[q01 + off]
              + f10 * xa[q10 + off] + f11 * xa[q11 + off];
      dst[r * 32 + kb + i] = f2bf(s);
    }
  };

  v8f acc[16] = {};

  if (tid < 32) tdm_copy_b(bw, &ldsB[0][0], 4096);
  buildA(0, &ldsA[0][0]);

  for (int kc = 0; kc < NKC; ++kc) {
    __builtin_amdgcn_s_wait_tensorcnt(0);
    __syncthreads();
    const int cur = kc & 1;
    if (kc + 1 < NKC) {
      if (tid < 32)
        tdm_copy_b(bw + (size_t)(kc + 1) * 8192, &ldsB[cur ^ 1][0], 4096);
      buildA(kc + 1, &ldsA[cur ^ 1][0]);
    }
    const __bf16* curA = &ldsA[cur][0];
    const __bf16* curB = &ldsB[cur][0];
    v8bf a0 = *(const v8bf*)&curA[(wave * 16 + l16) * 32 + half * 16];
    v8bf a1 = *(const v8bf*)&curA[(wave * 16 + l16) * 32 + half * 16 + 8];
    v16bf a = __builtin_shufflevector(a0, a1, 0,1,2,3,4,5,6,7,8,9,10,11,12,13,14,15);
    v16bf bcur = *(const v16bf*)&curB[lane * 16];
    #pragma unroll
    for (int n = 0; n < 16; ++n) {
      v16bf bnext = bcur;
      if (n < 15) bnext = *(const v16bf*)&curB[((n + 1) * 32 + lane) * 16];
      acc[n] = __builtin_amdgcn_wmma_f32_16x16x32_bf16(false, a, false, bcur,
                                                       (short)0, acc[n], false, false);
      bcur = bnext;
    }
  }
  #pragma unroll
  for (int n = 0; n < 16; ++n) {
    int o = n * 16 + l16;
    #pragma unroll
    for (int g = 0; g < 8; ++g) {
      int ml = wave * 16 + g + 8 * half;
      int mm = m0 + ml;
      int b2 = mm / HWn, hw2 = mm % HWn;
      out[((size_t)b2 * COUT + o) * HWn + hw2] = acc[n][g];
    }
  }
}

extern "C" void kernel_launch(void* const* d_in, const int* in_sizes, int n_in,
                              void* d_out, int out_size, void* d_ws, size_t ws_size,
                              hipStream_t stream) {
  (void)in_sizes; (void)n_in; (void)out_size; (void)ws_size;
  const float* x        = (const float*)d_in[0];
  const float* w_offset = (const float*)d_in[1];
  const float* b_offset = (const float*)d_in[2];
  const float* w_conv   = (const float*)d_in[3];
  float* out = (float*)d_out;

  char* ws = (char*)d_ws;
  size_t off = 0;
  float* om  = (float*)(ws + off); off += (size_t)MTOT * NOFFP * sizeof(float);
  float* w00 = (float*)(ws + off); off += (size_t)NPTS * sizeof(float);
  float* w01 = (float*)(ws + off); off += (size_t)NPTS * sizeof(float);
  float* w10 = (float*)(ws + off); off += (size_t)NPTS * sizeof(float);
  float* w11 = (float*)(ws + off); off += (size_t)NPTS * sizeof(float);
  int*   o00 = (int*)  (ws + off); off += (size_t)NPTS * sizeof(int);
  int*   o01 = (int*)  (ws + off); off += (size_t)NPTS * sizeof(int);
  int*   o10 = (int*)  (ws + off); off += (size_t)NPTS * sizeof(int);
  int*   o11 = (int*)  (ws + off); off += (size_t)NPTS * sizeof(int);
  __bf16* pwc = (__bf16*)(ws + off); off += (size_t)KDIM * COUT  * sizeof(__bf16);
  __bf16* pwo = (__bf16*)(ws + off); off += (size_t)KDIM * NOFFP * sizeof(__bf16);

  pack_wconv_k<<<(KDIM * COUT  + 255) / 256, 256, 0, stream>>>(w_conv, pwc);
  pack_woff_k <<<(KDIM * NOFFP + 255) / 256, 256, 0, stream>>>(w_offset, pwo);
  offset_gemm_k<<<MTOT / 128, 256, 0, stream>>>(x, pwo, b_offset, om);
  precomp_k<<<(NPTS + 255) / 256, 256, 0, stream>>>(om, w00, w01, w10, w11,
                                                    o00, o01, o10, o11);
  main_gemm_k<<<MTOT / 128, 256, 0, stream>>>(x, pwc, w00, w01, w10, w11,
                                              o00, o01, o10, o11, out);
}
